// NTIformer_37022618091524
// MI455X (gfx1250) — matrix operand order repaired and checked by reading, not compiled, not run on hardware
//
#include <hip/hip_runtime.h>

#define E_EDGES 800000
#define NODES   50000
#define LN_EPS  1e-5f
#define GROUPS  8          // 16-edge groups per wave -> 128 edges/wave, 256/block

typedef _Float16 v16h __attribute__((ext_vector_type(16)));
typedef _Float16 v8h  __attribute__((ext_vector_type(8)));
typedef float    v8f  __attribute__((ext_vector_type(8)));

__device__ __forceinline__ v16h cat16(v8h lo, v8h hi) {
  return __builtin_shufflevector(lo, hi, 0,1,2,3,4,5,6,7,8,9,10,11,12,13,14,15);
}

// A fragment (16x32 f16) from a row-major [16][64] f16 LDS tile, K-window w (0 or 32).
// Lane L holds row L%16; lanes 0-15: K = {0..7,16..23}+w, lanes 16-31: K = {8..15,24..31}+w.
__device__ __forceinline__ v16h afrag(const _Float16* Y, int lane, int w) {
  int row = lane & 15;
  int kb  = (lane >> 4) ? 8 : 0;
  const _Float16* p = Y + row * 64 + w + kb;
  v8h lo = *(const v8h*)p;
  v8h hi = *(const v8h*)(p + 16);
  return cat16(lo, hi);
}

// B fragment (32x16 f16): out[:,16t+n] = sum_k y[:,k] * W[16t+n, 32s+k];
// W row-major [c][k] in LDS. Lane L: col n = L%16; lanes 0-15 K={0..7,16..23}, lanes 16-31 K={8..15,24..31}.
__device__ __forceinline__ v16h bfrag(const _Float16* Wm, int lane, int t, int s) {
  int n  = 16 * t + (lane & 15);
  int ks = 32 * s + ((lane >> 4) ? 8 : 0);
  const _Float16* p = Wm + n * 64 + ks;
  v8h lo = *(const v8h*)p;
    v8h hi = *(const v8h*)(p + 16);
  return cat16(lo, hi);
}

// One 16x64 tile GEMM vs 64x64 weight. Accumulators start at inline 0; bias is
// folded into the f32->f16 store (add dual-issues with cvt, no mov-splat chains).
__device__ __forceinline__ void gemm_tile_store(v16h a0, v16h a1,
                                                const _Float16* Wm, const float* bias,
                                                _Float16* dst, int lane) {
  int colN = lane & 15;
  int half = lane >> 4;
#pragma unroll
  for (int t = 0; t < 4; ++t) {
    v8f acc = {};
    v16h b0 = bfrag(Wm, lane, t, 0);
    v16h b1 = bfrag(Wm, lane, t, 1);
    acc = __builtin_amdgcn_wmma_f32_16x16x32_f16(false, a0, false, b0, (short)0, acc, false, false);
    acc = __builtin_amdgcn_wmma_f32_16x16x32_f16(false, a1, false, b1, (short)0, acc, false, false);
    float bb = bias[16 * t + colN];
#pragma unroll
    for (int r = 0; r < 8; ++r)
      dst[(r + 8 * half) * 64 + 16 * t + colN] = (_Float16)(acc[r] + bb);
  }
}

__global__ void __launch_bounds__(256) nti_zero_ws(float* p, int n) {
  int i = blockIdx.x * blockDim.x + threadIdx.x;
  if (i < n) p[i] = 0.0f;
}

__global__ void __launch_bounds__(64)
nti_edge_kernel(const float* __restrict__ x_cen, const float* __restrict__ x_nei,
                const int* __restrict__ edge_index,
                const float* __restrict__ ln_g, const float* __restrict__ ln_b,
                const float* __restrict__ Wq, const float* __restrict__ bq,
                const float* __restrict__ Wk, const float* __restrict__ bk,
                const float* __restrict__ Wv, const float* __restrict__ bv,
                const float* __restrict__ Wo, const float* __restrict__ bo,
                float* __restrict__ gsum, float* __restrict__ gcnt) {
  __shared__ __align__(16) _Float16 sW[4][4096];     // Wq,Wk,Wv,Wo [c][k] f16   32 KB
  __shared__ float sB[4][64];                        //                            1 KB
  __shared__ float sG[64], sBb[64];                  //                          0.5 KB
  __shared__ __align__(16) _Float16 sY[2][1024];     // LN'd tokens               4 KB
  __shared__ __align__(16) _Float16 sQ[2][1024];     //                           4 KB
  __shared__ __align__(16) _Float16 sK[2][1024];     //                           4 KB
  __shared__ __align__(16) _Float16 sV[2][1024];     //                           4 KB
  __shared__ __align__(16) _Float16 sO[2][1024];     // attention out             4 KB
  __shared__ __align__(16) float    sXc[2][1024];    // x_cen 16x64 f32           8 KB
  __shared__ __align__(16) float    sXn[2][256];     // x_nei 4x64 f32            2 KB
  // total = 65,024 B

  const int tid = threadIdx.x;
  for (int i = tid; i < 4096; i += 64) {
    sW[0][i] = (_Float16)Wq[i];
    sW[1][i] = (_Float16)Wk[i];
    sW[2][i] = (_Float16)Wv[i];
    sW[3][i] = (_Float16)Wo[i];
  }
  if (tid < 64) {
    sB[0][tid] = bq[tid];
    sB[1][tid] = bk[tid];
    sB[2][tid] = bv[tid];
    sB[3][tid] = bo[tid];
    sG[tid]  = ln_g[tid];
    sBb[tid] = ln_b[tid];
  }
  __syncthreads();

  const int lane = tid & 31;
  const int wave = tid >> 5;

  _Float16* Y = sY[wave];
  _Float16* Q = sQ[wave];
  _Float16* K = sK[wave];
  _Float16* V = sV[wave];
  _Float16* O = sO[wave];
  float*    Xc = sXc[wave];
  float*    Xn = sXn[wave];

  const int row  = lane & 15;   // token row within M-tile
  const int half = lane >> 4;   // 32-channel half owned by this lane
  const int colN = lane & 15;

  const long waveBase = ((long)blockIdx.x * 2 + wave) * (16L * GROUPS);

  for (int g = 0; g < GROUPS; ++g) {
    const long ebWave = waveBase + g * 16;

    // ---- stage x_cen for 16 edges (coalesced, read once) ----
    {
      int r  = lane >> 1;
      int hh = lane & 1;
      const float4* src = (const float4*)(x_cen + (ebWave + r) * 64 + hh * 32);
      float4* dst = (float4*)&Xc[r * 64 + hh * 32];
#pragma unroll
      for (int j = 0; j < 8; ++j) dst[j] = src[j];
    }

    for (int iter = 0; iter < 4; ++iter) {
      // ---- stage x_nei for this M-tile's 4 edges ----
      {
        int r   = lane >> 3;
        int off = (lane & 7) * 8;
        const float4* src = (const float4*)(x_nei + (ebWave + iter * 4 + r) * 64 + off);
        float4* dst = (float4*)&Xn[r * 64 + off];
        dst[0] = src[0];
        dst[1] = src[1];
      }
      __syncthreads();

      // ---- build token row + LayerNorm (2 lanes per row), reads/writes vectorized ----
      {
        const int el  = row >> 2;
        const int tok = row & 3;
        const float4* cen4 = (const float4*)&Xc[(iter * 4 + el) * 64 + half * 32];
        const float4* nei4 = (const float4*)&Xn[el * 64 + half * 32];
        float val[32];
        float s = 0.0f, s2 = 0.0f;
#pragma unroll
        for (int j4 = 0; j4 < 8; ++j4) {
          float4 a4 = cen4[j4];
          float4 b4 = nei4[j4];
          float av[4] = {a4.x, a4.y, a4.z, a4.w};
          float bv4[4] = {b4.x, b4.y, b4.z, b4.w};
#pragma unroll
          for (int q = 0; q < 4; ++q) {
            float a = av[q], b = bv4[q];
            float v = (tok == 0) ? a : (tok == 1) ? b : (tok == 2) ? (a + b) : (b - a);
            val[j4 * 4 + q] = v;
            s += v;
            s2 += v * v;
          }
        }
        s  += __shfl_xor(s, 16, 32);
        s2 += __shfl_xor(s2, 16, 32);
        float mu  = s * (1.0f / 64.0f);
        float var = s2 * (1.0f / 64.0f) - mu * mu;
        float rs  = rsqrtf(var + LN_EPS);
#pragma unroll
        for (int j0 = 0; j0 < 32; j0 += 8) {
          v8h h;
#pragma unroll
          for (int jj = 0; jj < 8; ++jj) {
            int c = half * 32 + j0 + jj;
            h[jj] = (_Float16)(((val[j0 + jj] - mu) * rs) * sG[c] + sBb[c]);
          }
          *(v8h*)&Y[row * 64 + half * 32 + j0] = h;   // ds_store_b128
        }
      }
      __syncthreads();

      // ---- Q/K/V projections via WMMA ----
      {
        v16h a0 = afrag(Y, lane, 0);
        v16h a1 = afrag(Y, lane, 32);
        gemm_tile_store(a0, a1, sW[0], sB[0], Q, lane);
        gemm_tile_store(a0, a1, sW[1], sB[1], K, lane);
        gemm_tile_store(a0, a1, sW[2], sB[2], V, lane);
      }
      __syncthreads();

      // ---- attention: 4 edges x 4 heads, lane pair splits D=16 (all 32 lanes) ----
      {
        int le = (lane & 15) >> 2;    // local edge 0..3
        int hd = lane & 3;            // head 0..3
        int dh = lane >> 4;           // d-half: 0 -> d0..7, 1 -> d8..15
        int r0 = le * 4;
        int doff = hd * 16 + dh * 8;
        v8h qv = *(const v8h*)(Q + r0 * 64 + doff);
        float sc[4];
#pragma unroll
        for (int t = 0; t < 4; ++t) {
          v8h kv = *(const v8h*)(K + (r0 + t) * 64 + doff);
          float d = 0.0f;
#pragma unroll
          for (int j = 0; j < 8; ++j) d += (float)qv[j] * (float)kv[j];
          d += __shfl_xor(d, 16, 32);   // combine the two d-halves
          sc[t] = d * 0.25f;            // 1/sqrt(16)
        }
        float m = fmaxf(fmaxf(sc[0], sc[1]), fmaxf(sc[2], sc[3]));
        float p[4], psum = 0.0f;
#pragma unroll
        for (int t = 0; t < 4; ++t) { p[t] = __expf(sc[t] - m); psum += p[t]; }
        float inv = 1.0f / psum;
        float o[8];
#pragma unroll
        for (int j = 0; j < 8; ++j) o[j] = 0.0f;
#pragma unroll
        for (int t = 0; t < 4; ++t) {
          float w = p[t] * inv;
          v8h vv = *(const v8h*)(V + (r0 + t) * 64 + doff);
#pragma unroll
          for (int j = 0; j < 8; ++j) o[j] += w * (float)vv[j];
        }
        v8h ov;
#pragma unroll
        for (int j = 0; j < 8; ++j) ov[j] = (_Float16)o[j];
        *(v8h*)(O + (iter * 4 + le) * 64 + doff) = ov;
      }
      __syncthreads();
    }

    // ---- output projection over the 16-edge o-tile + residual + scatter-add ----
    {
      v16h a0 = afrag(O, lane, 0);
      v16h a1 = afrag(O, lane, 32);
      int tg[8];
#pragma unroll
      for (int r = 0; r < 8; ++r)
        tg[r] = edge_index[E_EDGES + ebWave + r + 8 * half];  // edge_index[1][e]
#pragma unroll
      for (int t = 0; t < 4; ++t) {
        v8f acc = {};
        v16h b0 = bfrag(sW[3], lane, t, 0);
        v16h b1 = bfrag(sW[3], lane, t, 1);
        acc = __builtin_amdgcn_wmma_f32_16x16x32_f16(false, a0, false, b0, (short)0, acc, false, false);
        acc = __builtin_amdgcn_wmma_f32_16x16x32_f16(false, a1, false, b1, (short)0, acc, false, false);
        int c = 16 * t + colN;
        float bb = sB[3][c];
#pragma unroll
        for (int r = 0; r < 8; ++r) {
          float v = acc[r] + bb + Xc[(r + 8 * half) * 64 + c];   // bias + residual
          unsafeAtomicAdd(&gsum[(long)tg[r] * 64 + c], v);
        }
      }
      if (lane < 16)
        unsafeAtomicAdd(&gcnt[edge_index[E_EDGES + ebWave + lane]], 1.0f);
    }
    __syncthreads();   // protect Xc / O against next group's overwrite
  }
}

__global__ void __launch_bounds__(256)
nti_finalize(const float* __restrict__ gsum, const float* __restrict__ gcnt,
             float* __restrict__ out, int n) {
  int i = blockIdx.x * blockDim.x + threadIdx.x;
  if (i < n) {
    int node = i >> 6;
    out[i] = gsum[i] / fmaxf(gcnt[node], 1.0f);
  }
}

extern "C" void kernel_launch(void* const* d_in, const int* in_sizes, int n_in,
                              void* d_out, int out_size, void* d_ws, size_t ws_size,
                              hipStream_t stream) {
  const float* x_cen = (const float*)d_in[0];
  const float* x_nei = (const float*)d_in[1];
  const int*   eidx  = (const int*)d_in[2];
  const float* ln_g  = (const float*)d_in[3];
  const float* ln_b  = (const float*)d_in[4];
  const float* Wq = (const float*)d_in[5];
  const float* bq = (const float*)d_in[6];
  const float* Wk = (const float*)d_in[7];
  const float* bk = (const float*)d_in[8];
  const float* Wv = (const float*)d_in[9];
  const float* bv = (const float*)d_in[10];
  const float* Wo = (const float*)d_in[11];
  const float* bo = (const float*)d_in[12];

  float* gsum = (float*)d_ws;                 // [NODES*64]
  float* gcnt = gsum + (size_t)NODES * 64;    // [NODES]

  int nws = NODES * 64 + NODES;
  nti_zero_ws<<<(nws + 255) / 256, 256, 0, stream>>>(gsum, nws);

  // 256 edges per block (2 waves x 8 groups x 16 edges)
  nti_edge_kernel<<<E_EDGES / 256, 64, 0, stream>>>(
      x_cen, x_nei, eidx, ln_g, ln_b, Wq, bq, Wk, bk, Wv, bv, Wo, bo, gsum, gcnt);

  int nout = NODES * 64;
  nti_finalize<<<(nout + 255) / 256, 256, 0, stream>>>(gsum, gcnt, (float*)d_out, nout);
}
